// RNN_1030792151246
// MI455X (gfx1250) — matrix operand order repaired and checked by reading, not compiled, not run on hardware
//
#include <hip/hip_runtime.h>
#include <hip/hip_bf16.h>

// SEQ=512, B=64, IN=H=1024, K = IN+H = 2048
// Per step: h0n = tanh([x,h0] @ W0.T + b0); h1n = tanh([h0n,h1] @ W1.T + b1)
// Hidden states are kept in bf16 (they only feed WMMA A-fragments);
// Y output and accumulation stay f32.

typedef __attribute__((ext_vector_type(16))) __bf16 v16bf;
typedef __attribute__((ext_vector_type(8)))  __bf16 v8bf;
typedef __attribute__((ext_vector_type(8)))  float  v8f;
typedef __attribute__((ext_vector_type(4)))  float  v4f;
typedef __attribute__((ext_vector_type(4)))  int    v4i;

#define HDIM   1024
#define KDIM   2048
#define BATCH  64
#define SEQLEN 512

#if __has_builtin(__builtin_amdgcn_global_load_async_to_lds_b128)
#define HAVE_ASYNC_LDS 1
#else
#define HAVE_ASYNC_LDS 0
#endif

// ---------------------------------------------------------------------------
// Pack W (f32, row-major [1024][2048]; Y = act @ W.T so B(k,n) = W[n][k])
// into bf16 WMMA-B-fragment tile order:
//   pW[((n_tile*64 + k_tile)*32 + lane)*16 + e] =
//       bf16( W[n_tile*16 + (lane&15)][k_tile*32 + (lane>>4)*8 + (e&7) + (e>>3)*16] )
// ---------------------------------------------------------------------------
__global__ void pack_w_kernel(const float* __restrict__ W,
                              __bf16* __restrict__ pW, int total) {
  int idx = blockIdx.x * blockDim.x + threadIdx.x;
  if (idx >= total) return;
  int e    = idx & 15;
  int lane = (idx >> 4) & 31;
  int kt   = (idx >> 9) & 63;
  int nt   = idx >> 15;
  int row  = nt * 16 + (lane & 15);
  int col  = kt * 32 + (lane >> 4) * 8 + (e & 7) + (e >> 3) * 16;
  pW[idx] = (__bf16)W[(size_t)row * KDIM + col];
}

__global__ void zero_f32_kernel(float* __restrict__ p, int n) {
  int i = blockIdx.x * blockDim.x + threadIdx.x;
  if (i < n) p[i] = 0.0f;
}

// ---------------------------------------------------------------------------
// A-fragment loaders (16x32 A layout: lane<16 -> M=lane, K in {0-7,16-23};
// lane>=16 -> M=lane-16, K in {8-15,24-31}).
// ---------------------------------------------------------------------------
__device__ __forceinline__ v16bf load_a_f32(const float* __restrict__ row,
                                            int kbase, int khalf) {
  const float* p0 = row + kbase + khalf * 8;
  const float* p1 = p0 + 16;
  v4f r0 = *(const v4f*)(p0);
  v4f r1 = *(const v4f*)(p0 + 4);
  v4f r2 = *(const v4f*)(p1);
  v4f r3 = *(const v4f*)(p1 + 4);
  __builtin_prefetch(p0 + 64, 0, 3);
  v16bf a;
#pragma unroll
  for (int i = 0; i < 4; ++i) {
    a[i]      = (__bf16)r0[i];
    a[i + 4]  = (__bf16)r1[i];
    a[i + 8]  = (__bf16)r2[i];
    a[i + 12] = (__bf16)r3[i];
  }
  return a;
}

__device__ __forceinline__ v16bf load_a_bf16(const __bf16* __restrict__ row,
                                             int kbase, int khalf) {
  const __bf16* p0 = row + kbase + khalf * 8;   // 16B aligned
  const __bf16* p1 = p0 + 16;
  v8bf r0 = *(const v8bf*)(p0);
  v8bf r1 = *(const v8bf*)(p1);
  v16bf a;
#pragma unroll
  for (int i = 0; i < 8; ++i) {
    a[i]     = r0[i];
    a[i + 8] = r1[i];
  }
  return a;
}

// Software-pipelined 32-step K loop; Loader(kbase) -> v16bf A fragment.
template <typename Loader>
__device__ __forceinline__ v8f k_loop(Loader ld, const __bf16* shB) {
  v8f acc = {};
  v16bf a = ld(0);
  v16bf b = *(const v16bf*)(shB);
#pragma unroll 4
  for (int kt = 0; kt < 31; ++kt) {
    v16bf an = ld((kt + 1) * 32);
    v16bf bn = *(const v16bf*)(shB + (kt + 1) * 512);
    acc = __builtin_amdgcn_wmma_f32_16x16x32_bf16(
        false, a, false, b, (short)0, acc, false, false);
    a = an;
    b = bn;
  }
  acc = __builtin_amdgcn_wmma_f32_16x16x32_bf16(
      false, a, false, b, (short)0, acc, false, false);
  return acc;
}

// ---------------------------------------------------------------------------
// One RNN layer for one timestep:
//   out = tanh( concat(A1[64][1024], A2[64][1024]) @ W.T + bias )
// A1 is f32 (layer 0 reads X) or bf16 (layer 1 reads h0n); A2 is always a
// bf16 hidden state. grid = 64 blocks (n_tile), 256 threads = 8 waves:
// wave = (kset<<2)|m_tile; kset 0 -> A1 K-half, kset 1 -> A2 K-half.
// 64KB weight strip staged to LDS via GLOBAL_LOAD_ASYNC_TO_LDS_B128; 32
// pipelined WMMAs per wave; K-partials reduced via LDS (buffer reuse).
// ---------------------------------------------------------------------------
template <bool A1_IS_F32>
__global__ __launch_bounds__(256) void rnn_layer_kernel(
    const void* __restrict__ A1v,     // [64][1024] f32 or bf16
    const __bf16* __restrict__ A2,    // [64][1024] bf16 hidden state
    const __bf16* __restrict__ pW,    // packed bf16 weights (tile order)
    const float* __restrict__ bias,   // [1024] f32
    __bf16* __restrict__ h_out,       // [64][1024] bf16 new hidden state
    float* __restrict__ y_out)        // optional [64][1024] f32 (layer 1)
{
  __shared__ __align__(64) __bf16 shW[64 * 512];   // 64KB: n_tile weight strip

  const int tid    = threadIdx.x;
  const int lane   = tid & 31;
  const int wave   = tid >> 5;
  const int m_tile = wave & 3;     // 0..3 -> batch rows
  const int kset   = wave >> 2;    // 0/1  -> K half
  const int khalf  = lane >> 4;
  const int mrow   = lane & 15;
  const int n_tile = blockIdx.x;   // 0..63

  // ---- stage this n_tile's 64KB weight strip into LDS ----
  const __bf16* gW = pW + (size_t)n_tile * (64 * 512);
#if HAVE_ASYNC_LDS
  {
    typedef __attribute__((address_space(1))) v4i gv4i;
    typedef __attribute__((address_space(3))) v4i lv4i;
    gv4i* gs = (gv4i*)gW;
    lv4i* ls = (lv4i*)shW;
#pragma unroll
    for (int r = 0; r < 16; ++r) {               // 256 thr * 16B * 16 = 64KB
      int off = r * 256 + tid;
      __builtin_amdgcn_global_load_async_to_lds_b128(gs + off, ls + off, 0, 0);
    }
#if __has_builtin(__builtin_amdgcn_s_wait_asynccnt)
    __builtin_amdgcn_s_wait_asynccnt(0);
#else
    asm volatile("s_wait_asynccnt 0x0" ::: "memory");
#endif
  }
#else
  {
    const uint4* gs = (const uint4*)gW;
    uint4* ls = (uint4*)shW;
#pragma unroll
    for (int r = 0; r < 16; ++r) ls[r * 256 + tid] = gs[r * 256 + tid];
  }
#endif
  __syncthreads();

  // ---- K loop: 32 WMMAs per wave, software pipelined ----
  const __bf16* shB = shW + (size_t)kset * (32 * 512) + lane * 16;
  const int rowidx = m_tile * 16 + mrow;

  v8f acc;
  if (kset == 0) {
    if constexpr (A1_IS_F32) {
      const float* arow = (const float*)A1v + (size_t)rowidx * HDIM;
      acc = k_loop([&](int kb) { return load_a_f32(arow, kb, khalf); }, shB);
    } else {
      const __bf16* arow = (const __bf16*)A1v + (size_t)rowidx * HDIM;
      acc = k_loop([&](int kb) { return load_a_bf16(arow, kb, khalf); }, shB);
    }
  } else {
    const __bf16* arow = A2 + (size_t)rowidx * HDIM;
    acc = k_loop([&](int kb) { return load_a_bf16(arow, kb, khalf); }, shB);
  }

  // ---- reduce the two K-halves via LDS (reuse shW after all B reads) ----
  __syncthreads();
  float* red = (float*)shW;                 // 4 tiles * 256 f32 = 4KB
  float* redp = red + m_tile * 256 + lane;
  if (kset == 1) {
#pragma unroll
    for (int r = 0; r < 8; ++r) redp[r * 32] = acc[r];
  }
  __syncthreads();
  if (kset == 0) {
#pragma unroll
    for (int r = 0; r < 8; ++r) acc[r] += redp[r * 32];

    // Epilogue: C/D layout -> N = lane%16, M = r + 8*khalf
    const int   n  = n_tile * 16 + (lane & 15);
    const float bv = bias[n];
#pragma unroll
    for (int r = 0; r < 8; ++r) {
      const int m = m_tile * 16 + khalf * 8 + r;
      const float v = tanhf(acc[r] + bv);
      h_out[(size_t)m * HDIM + n] = (__bf16)v;
      if (y_out) y_out[(size_t)m * HDIM + n] = v;
    }
  }
}

// ---------------------------------------------------------------------------
// Host driver. Workspace map (bytes):
//   [0,        4MiB)  pW0  bf16 packed
//   [4MiB,     8MiB)  pW1  bf16 packed
//   [8MiB, +256KiB)   h0   bf16 ping-pong (2 x 64*1024)
//   [...,  +256KiB)   h1   bf16 ping-pong (2 x 64*1024)
// ---------------------------------------------------------------------------
extern "C" void kernel_launch(void* const* d_in, const int* in_sizes, int n_in,
                              void* d_out, int out_size, void* d_ws,
                              size_t ws_size, hipStream_t stream) {
  const float* Xt = (const float*)d_in[0]; // [512][64][1024]
  const float* W0 = (const float*)d_in[1]; // [1024][2048]
  const float* b0 = (const float*)d_in[2]; // [1024]
  const float* W1 = (const float*)d_in[3]; // [1024][2048]
  const float* b1 = (const float*)d_in[4]; // [1024]
  float* Y = (float*)d_out;                // [512][64][1024]

  char* ws = (char*)d_ws;
  __bf16* pW0 = (__bf16*)(ws);
  __bf16* pW1 = (__bf16*)(ws + (size_t)4 * 1024 * 1024);
  __bf16* h0  = (__bf16*)(ws + (size_t)8 * 1024 * 1024);           // 2*65536 bf16
  __bf16* h1  = (__bf16*)(ws + (size_t)8 * 1024 * 1024 + 262144);  // 2*65536 bf16

  const int wtotal = HDIM * KDIM; // 2,097,152 elements per weight
  pack_w_kernel<<<(wtotal + 255) / 256, 256, 0, stream>>>(W0, pW0, wtotal);
  pack_w_kernel<<<(wtotal + 255) / 256, 256, 0, stream>>>(W1, pW1, wtotal);
  // zero all 4 bf16 hidden buffers: 4*65536*2B = 512KiB = 131072 f32 words
  zero_f32_kernel<<<(131072 + 255) / 256, 256, 0, stream>>>(
      (float*)(ws + (size_t)8 * 1024 * 1024), 131072);

  const size_t stepElems = (size_t)BATCH * HDIM; // 65536
  for (int t = 0; t < SEQLEN; ++t) {
    const float* x = Xt + (size_t)t * stepElems;
    __bf16* h0r = h0 + (size_t)(t & 1) * stepElems;
    __bf16* h0w = h0 + (size_t)((t + 1) & 1) * stepElems;
    __bf16* h1r = h1 + (size_t)(t & 1) * stepElems;
    __bf16* h1w = h1 + (size_t)((t + 1) & 1) * stepElems;

    // layer 0: h0n = tanh([x, h0] @ W0.T + b0)   (A1 = f32 X)
    rnn_layer_kernel<true><<<64, 256, 0, stream>>>(x, h0r, pW0, b0, h0w,
                                                   nullptr);
    // layer 1: h1n = tanh([h0n, h1] @ W1.T + b1); Y[t] = h1n  (A1 = bf16 h0n)
    rnn_layer_kernel<false><<<64, 256, 0, stream>>>(h0w, h1r, pW1, b1, h1w,
                                                    Y + (size_t)t * stepElems);
  }
}
